// LRULayer_90512140796633
// MI455X (gfx1250) — matrix operand order repaired and checked by reading, not compiled, not run on hardware
//
#include <hip/hip_runtime.h>
#include <stdint.h>

// ---------------- problem constants ----------------
constexpr int B_SZ = 4, L = 4096, H = 1024, P = 1024;
constexpr int M = B_SZ * L;            // 16384 rows (b*l)
constexpr int CHUNK = 256;             // scan chunk length
constexpr int NCHUNK = L / CHUNK;      // 16 chunks per sequence

// ---------------- WMMA types (CDNA5, wave32) ----------------
typedef __attribute__((ext_vector_type(16))) __bf16 v16bf;
typedef __attribute__((ext_vector_type(8)))  float  v8f;

union FragBF16 { uint4 u[2]; v16bf v; };

__device__ __forceinline__ uint16_t f2bf(float f) {
    uint32_t u = __float_as_uint(f);
    uint32_t r = u + 0x7FFFu + ((u >> 16) & 1u);   // round-to-nearest-even
    return (uint16_t)(r >> 16);
}

// ---------------- prep: Lambda + bf16(B_norm) ----------------
__global__ void k_lambda_wb(const float* __restrict__ nu_log,
                            const float* __restrict__ theta_log,
                            const float* __restrict__ B_re,
                            const float* __restrict__ B_im,
                            const float* __restrict__ gamma_log,
                            float* __restrict__ Lre, float* __restrict__ Lim,
                            uint16_t* __restrict__ WbRe, uint16_t* __restrict__ WbIm) {
    int p = blockIdx.x * blockDim.x + threadIdx.x;
    if (p >= P) return;
    float nu = expf(nu_log[p]);
    float th = expf(theta_log[p]);
    float r  = expf(-nu);
    Lre[p] = r * cosf(th);
    Lim[p] = r * sinf(th);
    float g = expf(gamma_log[p]);
    const size_t base = (size_t)p * H;
    for (int h = 0; h < H; ++h) {
        WbRe[base + h] = f2bf(B_re[base + h] * g);
        WbIm[base + h] = f2bf(B_im[base + h] * g);
    }
}

// ---------------- generic f32 -> bf16 convert ----------------
__global__ void k_f32_to_bf16(const float* __restrict__ src,
                              uint16_t* __restrict__ dst, int n) {
    int i = blockIdx.x * blockDim.x + threadIdx.x;
    if (i < n) dst[i] = f2bf(src[i]);
}

// ---------------- fragment loaders ----------------
// A fragment: 16x32 bf16 tile from row-major [rows][ldk] at (row0, k0).
// lane<16 holds row=lane&15, K = {0..7, 16..23}; lane>=16 holds K = {8..15, 24..31}.
__device__ __forceinline__ v16bf load_fragA(const uint16_t* base, int row0, int k0,
                                            int ldk, int lane) {
    int m   = lane & 15;
    int sel = lane >> 4;
    const uint16_t* pa = base + (size_t)(row0 + m) * ldk + k0 + sel * 8;
    FragBF16 f;
    f.u[0] = *(const uint4*)(pa);
    f.u[1] = *(const uint4*)(pa + 16);
    return f.v;
}

// B fragment: 32x16 bf16 tile from weights stored [N][K] row-major at (n0, k0).
// lane<16 holds col=lane, K = {0..7, 16..23}; lane>=16 holds K = {8..15, 24..31}.
__device__ __forceinline__ v16bf load_fragB(const uint16_t* base, int n0, int k0,
                                            int ldk, int lane) {
    int n   = lane & 15;
    int sel = lane >> 4;
    const uint16_t* pb = base + (size_t)(n0 + n) * ldk + k0 + sel * 8;
    FragBF16 f;
    f.u[0] = *(const uint4*)(pb);
    f.u[1] = *(const uint4*)(pb + 16);
    return f.v;
}

// ---------------- GEMM 1: Bu = inputs @ B_norm^T (re & im) ----------------
// A: Abf [M][H] bf16 ; W: [P][H] bf16 ; out: BuRe/BuIm [M][P] f32
__global__ __launch_bounds__(128)
void k_gemm_bu(const uint16_t* __restrict__ Abf,
               const uint16_t* __restrict__ WbRe,
               const uint16_t* __restrict__ WbIm,
               float* __restrict__ BuRe, float* __restrict__ BuIm) {
    int wave = threadIdx.x >> 5;
    int lane = threadIdx.x & 31;
    int m0 = blockIdx.x * 64 + (wave >> 1) * 32;
    int n0 = blockIdx.y * 64 + (wave & 1) * 32;
    int sel = lane >> 4;

    v8f accRe[2][2] = {};
    v8f accIm[2][2] = {};

    for (int k = 0; k < H; k += 32) {
        v16bf a[2], bre[2], bim[2];
        for (int mt = 0; mt < 2; ++mt)
            a[mt] = load_fragA(Abf, m0 + mt * 16, k, H, lane);
        for (int nt = 0; nt < 2; ++nt) {
            bre[nt] = load_fragB(WbRe, n0 + nt * 16, k, H, lane);
            bim[nt] = load_fragB(WbIm, n0 + nt * 16, k, H, lane);
        }
        for (int mt = 0; mt < 2; ++mt)
            for (int nt = 0; nt < 2; ++nt) {
                accRe[mt][nt] = __builtin_amdgcn_wmma_f32_16x16x32_bf16(
                    false, a[mt], false, bre[nt], (short)0, accRe[mt][nt], false, false);
                accIm[mt][nt] = __builtin_amdgcn_wmma_f32_16x16x32_bf16(
                    false, a[mt], false, bim[nt], (short)0, accIm[mt][nt], false, false);
            }
    }
    for (int mt = 0; mt < 2; ++mt)
        for (int nt = 0; nt < 2; ++nt) {
            int col   = n0 + nt * 16 + (lane & 15);
            int rbase = m0 + mt * 16 + sel * 8;
            for (int r = 0; r < 8; ++r) {
                size_t o = (size_t)(rbase + r) * P + col;
                BuRe[o] = accRe[mt][nt][r];
                BuIm[o] = accIm[mt][nt][r];
            }
        }
}

// ---------------- Scan stage 1: per-chunk local scan (in place) ----------------
__global__ void k_scan1(float* __restrict__ BuRe, float* __restrict__ BuIm,
                        const float* __restrict__ Lre, const float* __restrict__ Lim,
                        float* __restrict__ CarRe, float* __restrict__ CarIm) {
    int idx = blockIdx.x * blockDim.x + threadIdx.x;
    if (idx >= B_SZ * NCHUNK * P) return;
    int p = idx & (P - 1);
    int c = (idx / P) & (NCHUNK - 1);
    int b = idx / (P * NCHUNK);
    float lr = Lre[p], li = Lim[p];
    float xr = 0.f, xi = 0.f;
    size_t base = ((size_t)b * L + (size_t)c * CHUNK) * P + p;
    for (int t = 0; t < CHUNK; ++t) {
        size_t o = base + (size_t)t * P;
        float ur = BuRe[o], ui = BuIm[o];
        float nr = fmaf(lr, xr, fmaf(-li, xi, ur));
        float ni = fmaf(lr, xi, fmaf( li, xr, ui));
        xr = nr; xi = ni;
        BuRe[o] = xr; BuIm[o] = xi;
    }
    CarRe[idx] = xr; CarIm[idx] = xi;
}

// ---------------- Scan stage 2: cross-chunk prefix (exclusive) ----------------
__global__ void k_scan2(const float* __restrict__ Lre, const float* __restrict__ Lim,
                        float* __restrict__ CarRe, float* __restrict__ CarIm) {
    int idx = blockIdx.x * blockDim.x + threadIdx.x;
    if (idx >= B_SZ * P) return;
    int p = idx & (P - 1);
    int b = idx / P;
    float lr = Lre[p], li = Lim[p];
    float ar = lr, ai = li;                 // Lambda^CHUNK via 8 squarings (256 = 2^8)
    for (int s = 0; s < 8; ++s) {
        float nr = ar * ar - ai * ai;
        float ni = 2.f * ar * ai;
        ar = nr; ai = ni;
    }
    float xr = 0.f, xi = 0.f;               // exclusive prefix state
    for (int c = 0; c < NCHUNK; ++c) {
        size_t o = ((size_t)b * NCHUNK + c) * P + p;
        float cr = CarRe[o], ci = CarIm[o];
        CarRe[o] = xr; CarIm[o] = xi;       // state entering chunk c
        float nr = fmaf(ar, xr, fmaf(-ai, xi, cr));
        float ni = fmaf(ar, xi, fmaf( ai, xr, ci));
        xr = nr; xi = ni;
    }
}

// ---------------- Scan stage 3: fixup + bf16 convert (imag negated) ------------
__global__ void k_scan3(const float* __restrict__ BuRe, const float* __restrict__ BuIm,
                        const float* __restrict__ Lre, const float* __restrict__ Lim,
                        const float* __restrict__ CarRe, const float* __restrict__ CarIm,
                        uint16_t* __restrict__ XbfRe, uint16_t* __restrict__ XbfImN) {
    int idx = blockIdx.x * blockDim.x + threadIdx.x;
    if (idx >= B_SZ * NCHUNK * P) return;
    int p = idx & (P - 1);
    int c = (idx / P) & (NCHUNK - 1);
    int b = idx / (P * NCHUNK);
    float er = CarRe[idx], ei = CarIm[idx];       // carry entering this chunk
    float lr = Lre[p], li = Lim[p];
    float fr = lr, fi = li;                        // Lambda^(t+1)
    size_t base = ((size_t)b * L + (size_t)c * CHUNK) * P + p;
    for (int t = 0; t < CHUNK; ++t) {
        size_t o = base + (size_t)t * P;
        float cr = fr * er - fi * ei;
        float ci = fr * ei + fi * er;
        float xr = BuRe[o] + cr;
        float xi = BuIm[o] + ci;
        XbfRe[o]  = f2bf(xr);
        XbfImN[o] = f2bf(-xi);                     // pre-negated imaginary part
        float nfr = fr * lr - fi * li;
        float nfi = fr * li + fi * lr;
        fr = nfr; fi = nfi;
    }
}

// ---------------- GEMM 2: out = Xre@C_re^T + (-Xim)@C_im^T + D*u --------------
// X: [M][P] bf16 ; Wc: [H][P] bf16 ; U: [M][H] f32 ; Out: [M][H] f32
__global__ __launch_bounds__(128)
void k_gemm_out(const uint16_t* __restrict__ XRe,
                const uint16_t* __restrict__ XImN,
                const uint16_t* __restrict__ WcRe,
                const uint16_t* __restrict__ WcIm,
                const float* __restrict__ Dv,
                const float* __restrict__ U,
                float* __restrict__ Out) {
    int wave = threadIdx.x >> 5;
    int lane = threadIdx.x & 31;
    int m0 = blockIdx.x * 64 + (wave >> 1) * 32;
    int n0 = blockIdx.y * 64 + (wave & 1) * 32;
    int sel = lane >> 4;

    v8f acc[2][2] = {};

    for (int k = 0; k < P; k += 32) {
        v16bf are[2], aim[2], bre[2], bim[2];
        for (int mt = 0; mt < 2; ++mt) {
            are[mt] = load_fragA(XRe,  m0 + mt * 16, k, P, lane);
            aim[mt] = load_fragA(XImN, m0 + mt * 16, k, P, lane);
        }
        for (int nt = 0; nt < 2; ++nt) {
            bre[nt] = load_fragB(WcRe, n0 + nt * 16, k, P, lane);
            bim[nt] = load_fragB(WcIm, n0 + nt * 16, k, P, lane);
        }
        for (int mt = 0; mt < 2; ++mt)
            for (int nt = 0; nt < 2; ++nt) {
                acc[mt][nt] = __builtin_amdgcn_wmma_f32_16x16x32_bf16(
                    false, are[mt], false, bre[nt], (short)0, acc[mt][nt], false, false);
                acc[mt][nt] = __builtin_amdgcn_wmma_f32_16x16x32_bf16(
                    false, aim[mt], false, bim[nt], (short)0, acc[mt][nt], false, false);
            }
    }
    for (int mt = 0; mt < 2; ++mt)
        for (int nt = 0; nt < 2; ++nt) {
            int col   = n0 + nt * 16 + (lane & 15);
            int rbase = m0 + mt * 16 + sel * 8;
            float d = Dv[col];
            for (int r = 0; r < 8; ++r) {
                size_t row = (size_t)(rbase + r);
                Out[row * H + col] = acc[mt][nt][r] + d * U[row * H + col];
            }
        }
}

// ---------------- host launcher ----------------
extern "C" void kernel_launch(void* const* d_in, const int* in_sizes, int n_in,
                              void* d_out, int out_size, void* d_ws, size_t ws_size,
                              hipStream_t stream) {
    const float* inputs    = (const float*)d_in[0];
    const float* nu_log    = (const float*)d_in[1];
    const float* theta_log = (const float*)d_in[2];
    const float* B_re      = (const float*)d_in[3];
    const float* B_im      = (const float*)d_in[4];
    const float* C_re      = (const float*)d_in[5];
    const float* C_im      = (const float*)d_in[6];
    const float* Dv        = (const float*)d_in[7];
    const float* gamma_log = (const float*)d_in[8];

    char* ws = (char*)d_ws;
    size_t off = 0;
    auto alloc = [&](size_t bytes) -> void* {
        void* p = ws + off;
        off += (bytes + 255) & ~(size_t)255;
        return p;
    };

    uint16_t* Abf    = (uint16_t*)alloc((size_t)M * H * 2);      // bf16 inputs
    uint16_t* WbRe   = (uint16_t*)alloc((size_t)P * H * 2);
    uint16_t* WbIm   = (uint16_t*)alloc((size_t)P * H * 2);
    uint16_t* WcRe   = (uint16_t*)alloc((size_t)H * P * 2);
    uint16_t* WcIm   = (uint16_t*)alloc((size_t)H * P * 2);
    float*    BuRe   = (float*)   alloc((size_t)M * P * 4);
    float*    BuIm   = (float*)   alloc((size_t)M * P * 4);
    uint16_t* XbfRe  = (uint16_t*)alloc((size_t)M * P * 2);
    uint16_t* XbfImN = (uint16_t*)alloc((size_t)M * P * 2);
    float*    Lre    = (float*)   alloc((size_t)P * 4);
    float*    Lim    = (float*)   alloc((size_t)P * 4);
    float*    CarRe  = (float*)   alloc((size_t)B_SZ * NCHUNK * P * 4);
    float*    CarIm  = (float*)   alloc((size_t)B_SZ * NCHUNK * P * 4);
    (void)ws_size; (void)in_sizes; (void)n_in; (void)out_size;

    // prep
    k_lambda_wb<<<(P + 255) / 256, 256, 0, stream>>>(
        nu_log, theta_log, B_re, B_im, gamma_log, Lre, Lim, WbRe, WbIm);
    k_f32_to_bf16<<<(M * H) / 256, 256, 0, stream>>>(inputs, Abf, M * H);
    k_f32_to_bf16<<<(H * P) / 256, 256, 0, stream>>>(C_re, WcRe, H * P);
    k_f32_to_bf16<<<(H * P) / 256, 256, 0, stream>>>(C_im, WcIm, H * P);

    // GEMM 1: Bu (re, im)
    {
        dim3 grid(M / 64, P / 64);
        k_gemm_bu<<<grid, 128, 0, stream>>>(Abf, WbRe, WbIm, BuRe, BuIm);
    }

    // chunked linear-recurrence scan over L
    k_scan1<<<(B_SZ * NCHUNK * P) / 256, 256, 0, stream>>>(BuRe, BuIm, Lre, Lim, CarRe, CarIm);
    k_scan2<<<(B_SZ * P) / 256, 256, 0, stream>>>(Lre, Lim, CarRe, CarIm);
    k_scan3<<<(B_SZ * NCHUNK * P) / 256, 256, 0, stream>>>(
        BuRe, BuIm, Lre, Lim, CarRe, CarIm, XbfRe, XbfImN);

    // GEMM 2: out = Re(C x) + D*u
    {
        dim3 grid(M / 64, H / 64);
        k_gemm_out<<<grid, 128, 0, stream>>>(
            XbfRe, XbfImN, WcRe, WcIm, Dv, inputs, (float*)d_out);
    }
}
